// VariableSelectionNetwork_43525198577790
// MI455X (gfx1250) — compile-verified, matrix operand order checked by reading, not run on hardware
//
#include <hip/hip_runtime.h>
#include <hip/hip_bf16.h>

#define BQ   16
#define TQ   512
#define VQ   32
#define DQ   256
#define NTOK (BQ * TQ)          // 8192 tokens
#define PACK_PER_MAT (32 * 16 * 8 * 32 * 8)  // uints per packed weight matrix (4 MB)

typedef __attribute__((ext_vector_type(16))) __bf16 v16bf;
typedef __attribute__((ext_vector_type(8)))  float  v8f;

union Frag32 { uint4 q[2]; v16bf v; };

static __device__ __forceinline__ unsigned short f2bf(float f) {
  unsigned int u = __float_as_uint(f);
  u += 0x7fffu + ((u >> 16) & 1u);          // round-to-nearest-even
  return (unsigned short)(u >> 16);
}
static __device__ __forceinline__ float elu1(float x) {
  return x > 0.f ? x : (__expf(x) - 1.f);
}
static __device__ __forceinline__ float sigm(float x) {
  return 1.f / (1.f + __expf(-x));
}

// ---------------------------------------------------------------------------
// Pack w2/wa/wb [V][D][D] f32 into WMMA bf16 B-fragment layout:
//   pack[(((v*16 + nt16)*8 + kt)*32 + lane)*8 + jj]  (uint = 2 bf16)
//   column n = nt16*16 + (lane&15); lane<16 -> K {0..7,16..23}, lane>=16 -> +8
//   B[k][n] = w[v][n][k]  (contraction over last index e of 'vde')
// ---------------------------------------------------------------------------
__global__ __launch_bounds__(256)
void pack_b_kernel(const float* __restrict__ w2, const float* __restrict__ wa,
                   const float* __restrict__ wb,
                   unsigned int* __restrict__ w2p, unsigned int* __restrict__ wap,
                   unsigned int* __restrict__ wbp)
{
  int idx = blockIdx.x * blockDim.x + threadIdx.x;   // 0 .. 3*131072-1
  int which = idx >> 17;
  int r = idx & 131071;
  const float* w = (which == 0) ? w2 : (which == 1) ? wa : wb;
  unsigned int* o = (which == 0) ? w2p : (which == 1) ? wap : wbp;

  int lane = r & 31;
  int kt   = (r >> 5) & 7;
  int nt16 = (r >> 8) & 15;
  int v    = (r >> 12) & 31;

  int n   = nt16 * 16 + (lane & 15);
  int klo = (lane < 16) ? 0 : 8;
  size_t src = ((size_t)v * DQ + n) * DQ;
  size_t dst = (size_t)r * 8;

  #pragma unroll
  for (int jj = 0; jj < 8; ++jj) {
    int j0 = 2 * jj, j1 = 2 * jj + 1;
    int k0 = kt * 32 + klo + (j0 < 8 ? j0 : j0 + 8);
    int k1 = kt * 32 + klo + (j1 < 8 ? j1 : j1 + 8);
    unsigned int lo = f2bf(w[src + k0]);
    unsigned int hi = f2bf(w[src + k1]);
    o[dst + jj] = lo | (hi << 16);
  }
}

// ---------------------------------------------------------------------------
// Weight GRN + LayerNorm + softmax -> variable-selection weights [NT][V].
// One wave per token (V = 32 = wave32 lanes).
// ---------------------------------------------------------------------------
__global__ __launch_bounds__(256)
void grn_weights_kernel(const float* __restrict__ x,
                        const float* __restrict__ fc1w, const float* __restrict__ fc1b,
                        const float* __restrict__ fc2w, const float* __restrict__ fc2b,
                        const float* __restrict__ gaw,  const float* __restrict__ gab,
                        const float* __restrict__ gbw,  const float* __restrict__ gbb,
                        const float* __restrict__ glng, const float* __restrict__ glnb,
                        float* __restrict__ wout)
{
  __shared__ float g1s[8][DQ];
  int lane = threadIdx.x & 31;
  int wave = threadIdx.x >> 5;
  int tok  = blockIdx.x * 8 + wave;

  float xv = x[(size_t)tok * VQ + lane];

  // g1 = elu(x @ fc1w + fc1b)   [D], 8 entries per lane
  #pragma unroll
  for (int j = 0; j < 8; ++j) {
    int e = lane + 32 * j;
    float acc = fc1b[e];
    for (int vv = 0; vv < VQ; ++vv)
      acc += __shfl(xv, vv, 32) * fc1w[vv * DQ + e];
    g1s[wave][e] = elu1(acc);
  }
  __syncthreads();

  // g2 = g1 @ fc2w + fc2b  [V], lane owns column `lane`
  float g2 = fc2b[lane];
  for (int e = 0; e < DQ; ++e)
    g2 += g1s[wave][e] * fc2w[e * VQ + lane];

  float gav = gab[lane], gbv = gbb[lane];
  for (int k = 0; k < VQ; ++k) {
    float gk = __shfl(g2, k, 32);
    gav += gk * gaw[k * VQ + lane];
    gbv += gk * gbw[k * VQ + lane];
  }
  float y = xv + gav * sigm(gbv);

  float sum = y, sq = y * y;
  #pragma unroll
  for (int m = 1; m < 32; m <<= 1) {
    sum += __shfl_xor(sum, m, 32);
    sq  += __shfl_xor(sq,  m, 32);
  }
  float mu  = sum * (1.f / VQ);
  float var = sq * (1.f / VQ) - mu * mu;
  float logit = (y - mu) * rsqrtf(var + 1e-5f) * glng[lane] + glnb[lane];

  float mx = logit;
  #pragma unroll
  for (int m = 1; m < 32; m <<= 1) mx = fmaxf(mx, __shfl_xor(mx, m, 32));
  float ex = __expf(logit - mx);
  float es = ex;
  #pragma unroll
  for (int m = 1; m < 32; m <<= 1) es += __shfl_xor(es, m, 32);

  wout[(size_t)tok * VQ + lane] = ex / es;
}

// ---------------------------------------------------------------------------
// Fused VSN core. One workgroup = 64 tokens x full D x 16 variables.
// 16 waves: mi = wave>>2 (16-token group 0..3), ni = wave&3 (64-col strip).
// gridDim.x = 256: low 7 bits = token block (64 tokens), bit 7 = var half.
// Per variable:
//   h -> LDS(bf16) -> WMMA gemm1 -> h2 -> LDS(bf16) -> WMMA gemm2 (a,b)
//   -> GLU + skip -> LayerNorm(D) -> partial += weight * emb (registers).
// Partial outputs (per var-half) go to workspace; a reduce kernel sums them.
// ---------------------------------------------------------------------------
__global__ __launch_bounds__(512)
void vsn_main_kernel(const float* __restrict__ x,
                     const float* __restrict__ w1,    const float* __restrict__ b1,
                     const float* __restrict__ b2,
                     const float* __restrict__ ba,    const float* __restrict__ bb,
                     const float* __restrict__ wskip, const float* __restrict__ bskip,
                     const float* __restrict__ lng,   const float* __restrict__ lnb,
                     const unsigned int* __restrict__ w2p,
                     const unsigned int* __restrict__ wap,
                     const unsigned int* __restrict__ wbp,
                     const float* __restrict__ wsel,
                     float* __restrict__ part)        // [2][NT][D]
{
  __shared__ unsigned short hA[64][DQ + 8];   // bf16 tile (h, then h2), padded
  __shared__ float xs[64];
  __shared__ float wv[64];
  __shared__ float red[4][64][2];

  const int tid  = threadIdx.x;
  const int lane = tid & 31;
  const int wave = tid >> 5;
  const int mi   = wave >> 2;      // 0..3 : 16-token group
  const int ni   = wave & 3;       // 0..3 : 64-column strip
  const int l15  = lane & 15;
  const int half = (lane < 16) ? 0 : 1;
  const int tb   = blockIdx.x & 127;
  const int vh   = blockIdx.x >> 7;           // variable half (0/1)
  const int tok0 = tb * 64;
  const int v0   = vh * 16;

  const v8f vz = {0.f, 0.f, 0.f, 0.f, 0.f, 0.f, 0.f, 0.f};
  v8f oacc[4];
  #pragma unroll
  for (int t = 0; t < 4; ++t) oacc[t] = vz;

  const int hr  = tid >> 3;        // row for h-compute (0..63)
  const int hc0 = (tid & 7) * 32;  // col base for h-compute
  const int arow = mi * 16 + l15;  // A-fragment row
  const int klo  = half * 8;       // A/B fragment K sub-offset

  for (int v = v0; v < v0 + 16; ++v) {
    __syncthreads();
    if (tid < 64) {
      xs[tid] = x[(size_t)(tok0 + tid) * VQ + v];
      wv[tid] = wsel[(size_t)(tok0 + tid) * VQ + v];
    }
    { // h = elu(x * w1 + b1) -> LDS bf16
      float xval = x[(size_t)(tok0 + hr) * VQ + v];
      const float* w1v = w1 + v * DQ;
      const float* b1v = b1 + v * DQ;
      #pragma unroll 8
      for (int c = 0; c < 32; ++c) {
        int e = hc0 + c;
        hA[hr][e] = f2bf(elu1(xval * w1v[e] + b1v[e]));
      }
    }
    __syncthreads();

    // -------- GEMM1: h2 = h @ w2^T (per-var) --------
    v8f acc[4];
    #pragma unroll
    for (int t = 0; t < 4; ++t) acc[t] = vz;
    for (int kt = 0; kt < 8; ++kt) {
      Frag32 afr;
      afr.q[0] = *(const uint4*)&hA[arow][kt * 32 + klo];
      afr.q[1] = *(const uint4*)&hA[arow][kt * 32 + klo + 16];
      #pragma unroll
      for (int t = 0; t < 4; ++t) {
        int nt16 = ni * 4 + t;
        const uint4* bp = (const uint4*)(w2p +
            ((((size_t)v * 16 + nt16) * 8 + kt) * 32 + lane) * 8);
        Frag32 bfr; bfr.q[0] = bp[0]; bfr.q[1] = bp[1];
        acc[t] = __builtin_amdgcn_wmma_f32_16x16x32_bf16(
            false, afr.v, false, bfr.v, (short)0, acc[t], false, false);
      }
    }
    __syncthreads();
    // write h2 (+b2) back as bf16 A-source for gemm2
    #pragma unroll
    for (int t = 0; t < 4; ++t) {
      int ncol = (ni * 4 + t) * 16 + l15;
      float bias = b2[v * DQ + ncol];
      #pragma unroll
      for (int r = 0; r < 8; ++r) {
        int row = mi * 16 + r + half * 8;
        hA[row][ncol] = f2bf(acc[t][r] + bias);
      }
    }
    __syncthreads();

    // -------- GEMM2: a = h2@wa^T, b = h2@wb^T --------
    v8f acca[4], accb[4];
    #pragma unroll
    for (int t = 0; t < 4; ++t) { acca[t] = vz; accb[t] = vz; }
    for (int kt = 0; kt < 8; ++kt) {
      Frag32 afr;
      afr.q[0] = *(const uint4*)&hA[arow][kt * 32 + klo];
      afr.q[1] = *(const uint4*)&hA[arow][kt * 32 + klo + 16];
      #pragma unroll
      for (int t = 0; t < 4; ++t) {
        int nt16 = ni * 4 + t;
        size_t base = ((((size_t)v * 16 + nt16) * 8 + kt) * 32 + lane) * 8;
        const uint4* pa = (const uint4*)(wap + base);
        const uint4* pb = (const uint4*)(wbp + base);
        Frag32 bfa; bfa.q[0] = pa[0]; bfa.q[1] = pa[1];
        Frag32 bfb; bfb.q[0] = pb[0]; bfb.q[1] = pb[1];
        acca[t] = __builtin_amdgcn_wmma_f32_16x16x32_bf16(
            false, afr.v, false, bfa.v, (short)0, acca[t], false, false);
        accb[t] = __builtin_amdgcn_wmma_f32_16x16x32_bf16(
            false, afr.v, false, bfb.v, (short)0, accb[t], false, false);
      }
    }

    // -------- GLU + skip + z, per-row partial sums --------
    float s[8], s2[8];
    #pragma unroll
    for (int r = 0; r < 8; ++r) { s[r] = 0.f; s2[r] = 0.f; }
    #pragma unroll
    for (int t = 0; t < 4; ++t) {
      int ncol = (ni * 4 + t) * 16 + l15;
      float bav = ba[v * DQ + ncol];
      float bbv = bb[v * DQ + ncol];
      float wsk = wskip[v * DQ + ncol];
      float bsk = bskip[v * DQ + ncol];
      #pragma unroll
      for (int r = 0; r < 8; ++r) {
        int rloc = mi * 16 + r + half * 8;
        float av = acca[t][r] + bav;
        float bv = accb[t][r] + bbv;
        float z = xs[rloc] * wsk + bsk + av * sigm(bv);
        acca[t][r] = z;
        s[r] += z; s2[r] += z * z;
      }
    }
    // reduce over the 16 lanes of each half (same token row, different n)
    #pragma unroll
    for (int r = 0; r < 8; ++r) {
      #pragma unroll
      for (int m = 1; m < 16; m <<= 1) {
        s[r]  += __shfl_xor(s[r],  m, 32);
        s2[r] += __shfl_xor(s2[r], m, 32);
      }
    }
    if (l15 == 0) {
      #pragma unroll
      for (int r = 0; r < 8; ++r) {
        int rloc = mi * 16 + r + half * 8;
        red[ni][rloc][0] = s[r];
        red[ni][rloc][1] = s2[r];
      }
    }
    __syncthreads();

    // -------- LayerNorm over D, weighted accumulate --------
    #pragma unroll
    for (int t = 0; t < 4; ++t) {
      int ncol = (ni * 4 + t) * 16 + l15;
      float g  = lng[v * DQ + ncol];
      float bt = lnb[v * DQ + ncol];
      #pragma unroll
      for (int r = 0; r < 8; ++r) {
        int rloc = mi * 16 + r + half * 8;
        float tot  = red[0][rloc][0] + red[1][rloc][0] + red[2][rloc][0] + red[3][rloc][0];
        float tot2 = red[0][rloc][1] + red[1][rloc][1] + red[2][rloc][1] + red[3][rloc][1];
        float mu  = tot * (1.f / DQ);
        float var = tot2 * (1.f / DQ) - mu * mu;
        float sd  = fmaxf(sqrtf(fmaxf(var, 0.f)), 1e-5f);
        float emb = (acca[t][r] - mu) / sd * g + bt;
        oacc[t][r] += wv[rloc] * emb;
      }
    }
  }

  // -------- store partial [vh][NT][D] --------
  float* po = part + (size_t)vh * NTOK * DQ;
  #pragma unroll
  for (int t = 0; t < 4; ++t) {
    int ncol = (ni * 4 + t) * 16 + l15;
    #pragma unroll
    for (int r = 0; r < 8; ++r) {
      int rloc = mi * 16 + r + half * 8;
      po[(size_t)(tok0 + rloc) * DQ + ncol] = oacc[t][r];
    }
  }
}

// ---------------------------------------------------------------------------
// out = part[0] + part[1]  (float4 vectorized)
// ---------------------------------------------------------------------------
__global__ __launch_bounds__(256)
void reduce_out_kernel(const float* __restrict__ part, float* __restrict__ out)
{
  size_t i = (size_t)blockIdx.x * blockDim.x + threadIdx.x;  // over NT*D/4
  float4 a = ((const float4*)part)[i];
  float4 b = ((const float4*)(part + (size_t)NTOK * DQ))[i];
  float4 o;
  o.x = a.x + b.x; o.y = a.y + b.y; o.z = a.z + b.z; o.w = a.w + b.w;
  ((float4*)out)[i] = o;
}

// ---------------------------------------------------------------------------
extern "C" void kernel_launch(void* const* d_in, const int* in_sizes, int n_in,
                              void* d_out, int out_size, void* d_ws, size_t ws_size,
                              hipStream_t stream)
{
  const float* x     = (const float*)d_in[0];
  const float* w1    = (const float*)d_in[1];
  const float* b1    = (const float*)d_in[2];
  const float* w2    = (const float*)d_in[3];
  const float* b2    = (const float*)d_in[4];
  const float* wa    = (const float*)d_in[5];
  const float* ba    = (const float*)d_in[6];
  const float* wb    = (const float*)d_in[7];
  const float* bbv   = (const float*)d_in[8];
  const float* wskip = (const float*)d_in[9];
  const float* bskip = (const float*)d_in[10];
  const float* lng   = (const float*)d_in[11];
  const float* lnb   = (const float*)d_in[12];
  const float* gfc1w = (const float*)d_in[13];
  const float* gfc1b = (const float*)d_in[14];
  const float* gfc2w = (const float*)d_in[15];
  const float* gfc2b = (const float*)d_in[16];
  const float* gaw   = (const float*)d_in[17];
  const float* gab   = (const float*)d_in[18];
  const float* gbw   = (const float*)d_in[19];
  const float* gbb   = (const float*)d_in[20];
  const float* glng  = (const float*)d_in[21];
  const float* glnb  = (const float*)d_in[22];

  float* out  = (float*)d_out;                 // [NT][D]
  float* wsel = out + (size_t)NTOK * DQ;       // [NT][V] (2nd tuple output)

  unsigned int* w2p = (unsigned int*)d_ws;
  unsigned int* wap = w2p + (size_t)PACK_PER_MAT;
  unsigned int* wbp = wap + (size_t)PACK_PER_MAT;
  float* part = (float*)(wbp + (size_t)PACK_PER_MAT);   // [2][NT][D] = 16 MB

  pack_b_kernel<<<1536, 256, 0, stream>>>(w2, wa, wb, w2p, wap, wbp);
  grn_weights_kernel<<<NTOK / 8, 256, 0, stream>>>(
      x, gfc1w, gfc1b, gfc2w, gfc2b, gaw, gab, gbw, gbb, glng, glnb, wsel);
  vsn_main_kernel<<<256, 512, 0, stream>>>(
      x, w1, b1, b2, ba, bbv, wskip, bskip, lng, lnb, w2p, wap, wbp, wsel, part);
  reduce_out_kernel<<<(NTOK * DQ / 4) / 256, 256, 0, stream>>>(part, out);
}
